// DepthConditionedGWM_36464272343166
// MI455X (gfx1250) — matrix operand order repaired and checked by reading, not compiled, not run on hardware
//
#include <hip/hip_runtime.h>
#include <hip/hip_bf16.h>
#include <math.h>
#include <stdint.h>

typedef _Float16 half_t;
typedef __attribute__((ext_vector_type(16))) _Float16 v16h;
typedef __attribute__((ext_vector_type(8)))  _Float16 v8h;
typedef __attribute__((ext_vector_type(8)))  float    v8f;

#define NB    2048
#define NH    16
#define ND    64
#define NKMAX 32
#define NDE   32

// workspace layout (bytes)
#define WS_PQH 0        // 2048 half = 4096 B
#define WS_GLC 4096     // 64 f32
#define WS_GRC 4352     // 64 f32
#define WS_PR  4608     // 16 f32
#define WS_PI  4672     // 16 f32
#define WS_SA  4736     // 1 f32

__device__ __forceinline__ float sigmoidf_(float x) { return 1.f / (1.f + __expf(-x)); }

// ---------------- prep kernel: all depth-only constants ----------------
__global__ __launch_bounds__(256) void gwm_prep(
    const float* glw, const float* glb, const float* grw, const float* grb,
    const float* pqb, const float* qow, const float* qob,
    const float* skip_alpha, const float* wf, const float* wd, const float* wp,
    const int* depth_p,
    half_t* pq_h, float* glc, float* grc, float* prv, float* piv, float* sav)
{
  const int tid = threadIdx.x;
  const float depth = (float)depth_p[0];
  float dv[NDE];
#pragma unroll
  for (int i = 0; i < NDE / 2; ++i) {
    float inv = __expf((2.f * i) * (-logf(10000.f) / (float)NDE));
    float a = depth * inv;
    dv[2 * i]     = __sinf(a);
    dv[2 * i + 1] = __cosf(a);
  }
  // pq = pqb + dv @ qow + qob  (as fp16 for WMMA A operand)
  for (int o = tid; o < NKMAX * ND; o += 256) {
    float acc = pqb[o] + qob[o];
#pragma unroll
    for (int e = 0; e < NDE; ++e) acc += dv[e] * qow[e * (NKMAX * ND) + o];
    pq_h[o] = (half_t)acc;
  }
  // fold dv-slice of gate weights into biases
  if (tid < ND) {
    float a = glb[tid], b2 = grb[tid];
#pragma unroll
    for (int e = 0; e < NDE; ++e) {
      a  += dv[e] * glw[(2 * ND + e) * ND + tid];
      b2 += dv[e] * grw[(2 * ND + e) * ND + tid];
    }
    glc[tid] = a; grc[tid] = b2;
  }
  // per-head rotation scalars; 'scale' cancels algebraically
  if (tid < NH) {
    float sp  = log1pf(__expf(wd[tid]));       // softplus(wd) = alpha_d*scale
    float dec = __expf(-sp);                    // decay
    float ang = wf[tid] + wp[tid] + depth * 0.78539816339744831f; // omega*scale+phi
    prv[tid] = dec * __cosf(ang);
    piv[tid] = dec * __sinf(ang);
  }
  if (tid == 0) sav[0] = sigmoidf_(skip_alpha[0]);
}

// ---------------- per-wave shared state ----------------
struct WaveShm {
  union {
    float inbuf[2 * 16 * ND];   // f_left | rot(f_right)   (8 KB)
    float parent[32 * ND];      // reused for parent_raw   (8 KB)
  } u;
  half_t bank[32 * ND];         // fp16 bank for WMMA      (4 KB)
  float  Smat[32 * 33];         // scores (padded)         (4.2 KB)
  half_t P[32 * 32];            // softmax probs fp16      (2 KB)
  float lm[ND], rm[ND], gl[ND], gr[ND], sk[ND];
  float mu[32], rstd[32];
};

// ---- ISA 7.12.2 fragment builders (wave32) ----
// A (16x32 f16), contiguous k order used here (same as B, not ISA table order): lanes 0-15 rows 0-15 K{0..15}; lanes 16-31 same rows K{16..31}
__device__ __forceinline__ v16h load_a16(const half_t* M, int row, int stride, int koff, int hi) {
  v16h a;
#pragma unroll
  for (int i = 0; i < 4; ++i) {
    int k = koff + 2 * i + hi * 16;
    a[2 * i]     = M[row * stride + k];
    a[2 * i + 1] = M[row * stride + k + 1];
  }
#pragma unroll
  for (int i = 0; i < 4; ++i) {
    int k = koff + 8 + 2 * i + hi * 16;
    a[8 + 2 * i]     = M[row * stride + k];
    a[8 + 2 * i + 1] = M[row * stride + k + 1];
  }
  return a;
}
// B (32x16 f16) for S = pq @ bank^T : Bmat[d][n] = bank[n][d]
__device__ __forceinline__ v16h load_bT(const half_t* bank, int nrow, int koff, int hi) {
  v16h b;
#pragma unroll
  for (int i = 0; i < 8; ++i) {
    int d = koff + hi * 16 + 2 * i;
    b[2 * i]     = bank[nrow * ND + d];
    b[2 * i + 1] = bank[nrow * ND + d + 1];
  }
  return b;
}
// B (32x16 f16) for parent = P @ bank : Bmat[k][d] = bank[k][d]
__device__ __forceinline__ v16h load_bK(const half_t* bank, int ncol, int hi) {
  v16h b;
#pragma unroll
  for (int i = 0; i < 8; ++i) {
    int k = hi * 16 + 2 * i;
    b[2 * i]     = bank[k * ND + ncol];
    b[2 * i + 1] = bank[(k + 1) * ND + ncol];
  }
  return b;
}

#define WMMA_F16(a, b, c) \
  __builtin_amdgcn_wmma_f32_16x16x32_f16(false, (a), false, (b), (short)0, (c), false, false)

// low 32 bits of a generic pointer into __shared__ == LDS byte offset (ISA 10.2)
__device__ __forceinline__ uint32_t lds_off(const void* p) {
  return (uint32_t)(uintptr_t)p;
}

// ---------------- main kernel: one wave per (b,h) tile ----------------
__global__ __launch_bounds__(64) void gwm_main(
    const float* __restrict__ f_left, const float* __restrict__ f_right,
    const float* __restrict__ glw, const float* __restrict__ grw,
    const float* __restrict__ spw,
    const float* __restrict__ gamma, const float* __restrict__ beta,
    const half_t* __restrict__ pq_ws,
    const float* __restrict__ glc, const float* __restrict__ grc,
    const float* __restrict__ prv, const float* __restrict__ piv,
    const float* __restrict__ sav,
    float* __restrict__ out)
{
  __shared__ WaveShm shm[2];
  __shared__ half_t pqs[NKMAX * ND];

  const int tid  = threadIdx.x;
  const int w    = tid >> 5;
  const int lane = tid & 31;
  for (int i = tid; i < NKMAX * ND; i += 64) pqs[i] = pq_ws[i];

  const int t = blockIdx.x * 2 + w;           // flat (b*H + h)
  const int h = t & (NH - 1);
  WaveShm& ws = shm[w];
  float* fl = ws.u.inbuf;
  float* rr = ws.u.inbuf + 16 * ND;
  const float prh = prv[h], pih = piv[h];
  const float sa = sav[0];
  const size_t base = (size_t)t * (16 * ND);

  // ---- stage f_left via async global->LDS DMA (ASYNCcnt path) ----
  {
    const uint64_t gsrc = (uint64_t)(uintptr_t)(f_left + base);
    const uint32_t ldst = lds_off(fl);
#pragma unroll
    for (int j = 0; j < 8; ++j) {
      uint32_t off = (uint32_t)(lane + 32 * j) * 16u;
      uint32_t l = ldst + off;
      uint64_t g = gsrc + off;
      asm volatile("global_load_async_to_lds_b128 %0, %1, off"
                   :: "v"(l), "v"(g) : "memory");
    }
  }

  // rotate f_right while the DMA runs: rot = [pr*fr - pi*fi, pi*fr + pr*fi]
  const float* FR = f_right + base;
#pragma unroll
  for (int j = 0; j < 4; ++j) {
    int p = lane + 32 * j;          // pair index 0..127
    int row = p >> 3;
    int c = (p & 7) * 4;            // quad within low half
    const float4 fr = *(const float4*)(FR + row * ND + c);
    const float4 fi = *(const float4*)(FR + row * ND + c + 32);
    float4 lo, hq;
    lo.x = prh * fr.x - pih * fi.x; lo.y = prh * fr.y - pih * fi.y;
    lo.z = prh * fr.z - pih * fi.z; lo.w = prh * fr.w - pih * fi.w;
    hq.x = pih * fr.x + prh * fi.x; hq.y = pih * fr.y + prh * fi.y;
    hq.z = pih * fr.z + prh * fi.z; hq.w = pih * fr.w + prh * fi.w;
    *(float4*)(rr + row * ND + c)      = lo;
    *(float4*)(rr + row * ND + c + 32) = hq;
  }
  asm volatile("s_wait_asynccnt 0x0" ::: "memory");
  __syncthreads();

  // means over the 16 child slots, 2 dims per lane
#pragma unroll
  for (int s = 0; s < 2; ++s) {
    int d = lane + 32 * s;
    float sl = 0.f, sr = 0.f;
#pragma unroll
    for (int k = 0; k < 16; ++k) { sl += fl[k * ND + d]; sr += rr[k * ND + d]; }
    ws.lm[d] = sl * (1.f / 16.f);
    ws.rm[d] = sr * (1.f / 16.f);
  }
  __syncthreads();

  // gates (160->64 matvecs, dv part prefolded) + skip projection
#pragma unroll
  for (int s = 0; s < 2; ++s) {
    int d = lane + 32 * s;
    float al = glc[d], ar = grc[d], as_ = 0.f;
    for (int j = 0; j < ND; ++j) {
      float l = ws.lm[j], r = ws.rm[j];
      al  += l * glw[j * ND + d] + r * glw[(ND + j) * ND + d];
      ar  += l * grw[j * ND + d] + r * grw[(ND + j) * ND + d];
      as_ += l * spw[j * ND + d];
    }
    ws.gl[d] = sigmoidf_(al);
    ws.gr[d] = sigmoidf_(ar);
    ws.sk[d] = as_;
  }
  __syncthreads();

  // bank = [f_left*g_l ; rot*g_r] as fp16, one row per lane, 16B LDS stores
  {
    const int row = lane;
    const float* src = (row < 16) ? (fl + row * ND) : (rr + (row - 16) * ND);
    const float* g   = (row < 16) ? ws.gl : ws.gr;
#pragma unroll
    for (int c = 0; c < ND; c += 8) {
      v8h hv;
#pragma unroll
      for (int k = 0; k < 8; ++k) hv[k] = (half_t)(src[c + k] * g[c + k]);
      *(v8h*)(ws.bank + row * ND + c) = hv;
    }
  }
  __syncthreads();

  const int row15 = lane & 15;
  const int hi    = lane >> 4;

  // S = (pq @ bank^T) / sqrt(D): 2x2 tiles, 2 K-chunks of 32
  v8f s00 = {}, s01 = {}, s10 = {}, s11 = {};
#pragma unroll
  for (int ch = 0; ch < 2; ++ch) {
    int koff = ch * 32;
    v16h a0 = load_a16(pqs, row15, ND, koff, hi);
    v16h a1 = load_a16(pqs, 16 + row15, ND, koff, hi);
    v16h b0 = load_bT(ws.bank, row15, koff, hi);
    v16h b1 = load_bT(ws.bank, 16 + row15, koff, hi);
    s00 = WMMA_F16(a0, b0, s00);
    s01 = WMMA_F16(a0, b1, s01);
    s10 = WMMA_F16(a1, b0, s10);
    s11 = WMMA_F16(a1, b1, s11);
  }
#pragma unroll
  for (int i = 0; i < 8; ++i) {
    int r = hi * 8 + i;
    ws.Smat[r * 33 + row15]             = s00[i] * 0.125f;
    ws.Smat[r * 33 + 16 + row15]        = s01[i] * 0.125f;
    ws.Smat[(16 + r) * 33 + row15]      = s10[i] * 0.125f;
    ws.Smat[(16 + r) * 33 + 16 + row15] = s11[i] * 0.125f;
  }
  __syncthreads();

  // softmax per row (lane = row)
  {
    float m = -3.4e38f;
    for (int c = 0; c < 32; ++c) m = fmaxf(m, ws.Smat[lane * 33 + c]);
    float ssum = 0.f;
    for (int c = 0; c < 32; ++c) {
      float e = __expf(ws.Smat[lane * 33 + c] - m);
      ws.Smat[lane * 33 + c] = e;
      ssum += e;
    }
    float inv = 1.f / ssum;
    for (int c = 0; c < 32; ++c) ws.P[lane * 32 + c] = (half_t)(ws.Smat[lane * 33 + c] * inv);
  }
  __syncthreads();

  // parent_raw = P @ bank: 2x4 tiles, single K=32 WMMA each
  v16h pa0 = load_a16(ws.P, row15, 32, 0, hi);
  v16h pa1 = load_a16(ws.P, 16 + row15, 32, 0, hi);
#pragma unroll
  for (int dt = 0; dt < 4; ++dt) {
    int ncol = dt * 16 + row15;
    v16h bb = load_bK(ws.bank, ncol, hi);
    v8f d0 = {}, d1 = {};
    d0 = WMMA_F16(pa0, bb, d0);
    d1 = WMMA_F16(pa1, bb, d1);
#pragma unroll
    for (int i = 0; i < 8; ++i) {
      ws.u.parent[(hi * 8 + i) * ND + ncol]      = d0[i];
      ws.u.parent[(16 + hi * 8 + i) * ND + ncol] = d1[i];
    }
  }
  __syncthreads();

  // layernorm stats per row (lane = row)
  {
    float s1 = 0.f, s2 = 0.f;
    for (int d = 0; d < ND; ++d) {
      float x = ws.u.parent[lane * ND + d];
      s1 += x; s2 += x * x;
    }
    float mu  = s1 * (1.f / ND);
    float var = s2 * (1.f / ND) - mu * mu;
    ws.mu[lane]   = mu;
    ws.rstd[lane] = rsqrtf(var + 1e-5f);
  }
  __syncthreads();

  // normalize in place (float4 granularity): norm*gamma + beta + sigmoid(alpha)*skip
  float4* p4 = (float4*)ws.u.parent;
#pragma unroll
  for (int j = 0; j < 16; ++j) {
    int q = lane + 32 * j;        // 512 float4 chunks
    int row = q >> 4;
    int d  = (q & 15) * 4;
    float4 x = p4[q];
    float m = ws.mu[row], rs = ws.rstd[row];
    x.x = (x.x - m) * rs * gamma[d + 0] + beta[d + 0] + sa * ws.sk[d + 0];
    x.y = (x.y - m) * rs * gamma[d + 1] + beta[d + 1] + sa * ws.sk[d + 1];
    x.z = (x.z - m) * rs * gamma[d + 2] + beta[d + 2] + sa * ws.sk[d + 2];
    x.w = (x.w - m) * rs * gamma[d + 3] + beta[d + 3] + sa * ws.sk[d + 3];
    p4[q] = x;
  }

  // drain this wave's LDS writes, then async store LDS->global (ASYNCcnt path;
  // S_ENDPGM performs an implicit wait-idle so no trailing wait is needed)
  asm volatile("s_wait_dscnt 0x0" ::: "memory");
  {
    const uint64_t gdst = (uint64_t)(uintptr_t)(out + (size_t)t * (32 * ND));
    const uint32_t lsrc = lds_off(ws.u.parent);
#pragma unroll
    for (int j = 0; j < 16; ++j) {
      uint32_t off = (uint32_t)(lane + 32 * j) * 16u;
      uint64_t g = gdst + off;
      uint32_t l = lsrc + off;
      asm volatile("global_store_async_from_lds_b128 %0, %1, off"
                   :: "v"(g), "v"(l) : "memory");
    }
  }
}

extern "C" void kernel_launch(void* const* d_in, const int* in_sizes, int n_in,
                              void* d_out, int out_size, void* d_ws, size_t ws_size,
                              hipStream_t stream) {
  const float* f_left     = (const float*)d_in[0];
  const float* f_right    = (const float*)d_in[1];
  const float* glw        = (const float*)d_in[2];
  const float* glb        = (const float*)d_in[3];
  const float* grw        = (const float*)d_in[4];
  const float* grb        = (const float*)d_in[5];
  const float* pqb        = (const float*)d_in[6];
  const float* qow        = (const float*)d_in[7];
  const float* qob        = (const float*)d_in[8];
  const float* ln_gamma   = (const float*)d_in[9];
  const float* ln_beta    = (const float*)d_in[10];
  const float* spw        = (const float*)d_in[11];
  const float* skip_alpha = (const float*)d_in[12];
  const float* wf         = (const float*)d_in[13];
  const float* wd         = (const float*)d_in[14];
  const float* wp         = (const float*)d_in[15];
  const int*   depth      = (const int*)d_in[16];

  char* wsb = (char*)d_ws;
  half_t* pq_h = (half_t*)(wsb + WS_PQH);
  float*  glc  = (float*)(wsb + WS_GLC);
  float*  grc  = (float*)(wsb + WS_GRC);
  float*  prv  = (float*)(wsb + WS_PR);
  float*  piv  = (float*)(wsb + WS_PI);
  float*  sav  = (float*)(wsb + WS_SA);

  gwm_prep<<<1, 256, 0, stream>>>(glw, glb, grw, grb, pqb, qow, qob,
                                  skip_alpha, wf, wd, wp, depth,
                                  pq_h, glc, grc, prv, piv, sav);

  gwm_main<<<(NB * NH) / 2, 64, 0, stream>>>(
      f_left, f_right, glw, grw, spw, ln_gamma, ln_beta,
      pq_h, glc, grc, prv, piv, sav, (float*)d_out);
}
